// PNHead_UpScale_seg_27788438405908
// MI455X (gfx1250) — matrix operand order repaired and checked by reading, not compiled, not run on hardware
//
#include <hip/hip_runtime.h>
#include <hip/hip_bf16.h>
#include <math.h>

typedef __attribute__((ext_vector_type(16))) __bf16 v16bf;
typedef __attribute__((ext_vector_type(8)))  float  v8f;

#define NPTS 16384
#define NCTR 128

__device__ __constant__ int   c_ns[4]   = {4096, 1024, 256, 64};
__device__ __constant__ int   c_soff[4] = {0, 128*4096, 128*4096 + 128*1024,
                                           128*4096 + 128*1024 + 128*256};
__device__ __constant__ float c_r2[4]   = {0.01f, 0.16f, 0.64f, 2.56f};

// Fragment-region sizes (bf16 elements), all weights pre-swizzled into
// per-fragment lane-contiguous order: frag(kb,t): lane l holds 16 bf16
// with K = kb*32 + 16*(e>>3) + 8*(l>>4) + (e&7), N = t*16 + (l&15).
#define FR_SAW0 (4 * 1 * 8 * 512)    /* Kpad=32  N=128, per scale 8 frags  */
#define FR_SAW1 (4 * 4 * 8 * 512)    /* K=128    N=128                      */
#define FR_SAW2 (4 * 4 * 16 * 512)   /* K=128    N=256                      */
#define FR_FPW0 (32 * 8 * 512)       /* K=1024   N=128                      */
#define FR_FPW1 (4 * 4 * 512)        /* K=128    N=64                       */
#define FR_C1W  (2 * 4 * 512)        /* K=64     N=64                       */
#define FR_TOTAL (FR_SAW0 + FR_SAW1 + FR_SAW2 + FR_FPW0 + FR_FPW1 + FR_C1W)

// Per-scale LDS weight bytes for sa_mlp (dynamic shared): 104 KB.
#define SA_LDS_W_BYTES ((FR_SAW0 / 4 + FR_SAW1 / 4 + FR_SAW2 / 4) * 2)

// ---------------- WMMA helpers (bf16 -> f32, 16x16x32) ----------------

__device__ __forceinline__ v8f wmma_bf16(v16bf a, v16bf b, v8f c) {
  return __builtin_amdgcn_wmma_f32_16x16x32_bf16(false, a, false, b,
                                                 (short)0, c, false, false);
}

// Make an index opaque to the optimizer: blocks LICM from hoisting fragment
// loads (which previously spilled hundreds of VGPRs to scratch) while keeping
// pointer provenance (LDS loads stay ds_*, global loads stay global_*).
__device__ __forceinline__ int opaque(int v) {
  asm volatile("" : "+v"(v));
  return v;
}

// B fragment: one contiguous 32-byte load per lane from pre-swizzled weights.
__device__ __forceinline__ v16bf load_b_frag(const __bf16* Wf, int nt, int kb,
                                             int t, int lane) {
  int idx = opaque(((kb * nt + t) * 32 + lane) << 4);
  return *(const v16bf*)(Wf + idx);
}

// A fragment from LDS tile stored in A-swizzled layout:
// addr(m, k) = (k/32)*512 + m*32 + 16*h + e with h=(r>>3)&1,
// e=(r&7)|((r&16)>>1), r=k%32. Lane (h=l>>4, m=l&15) reads 16 contiguous bf16.
__device__ __forceinline__ v16bf load_a_frag(const __bf16* H, int kb, int lane) {
  return *(const v16bf*)(H + kb * 512 + (lane & 15) * 32 + ((lane >> 4) << 4));
}

// Swizzled offset within a 32-wide K block for channel r (0..31).
__device__ __forceinline__ int aswz(int r) {
  return (((r >> 3) & 1) << 4) | (r & 7) | ((r & 16) >> 1);
}

// Broadcast a scalar into a C/D fragment.
__device__ __forceinline__ v8f splat8(float bv) {
  v8f c;
#pragma unroll
  for (int j = 0; j < 8; ++j) c[j] = bv;
  return c;
}

// Broadcast per-column bias into a C/D fragment.
__device__ __forceinline__ v8f bias_frag(const float* b, int nbase, int lane) {
  return splat8(b[nbase + (lane & 15)]);
}

// Store C fragment (rows j+8*hl, column nbase+n) into A-swizzled LDS layout
// for the next layer, with ReLU.
__device__ __forceinline__ void store_c_relu_af(__bf16* H, int nbase, v8f c,
                                                int lane) {
  int hl = lane >> 4, n = lane & 15;
  int cidx = nbase + n;                 // output channel = next-layer K index
  __bf16* base = H + (cidx >> 5) * 512 + aswz(cidx & 31) + 8 * hl * 32;
#pragma unroll
  for (int j = 0; j < 8; ++j) {
    float v = c[j];
    v = v > 0.f ? v : 0.f;
    base[j * 32] = (__bf16)v;
  }
}

// ---------------- Tensor Data Mover (async weight staging) -------------------
#if __has_builtin(__builtin_amdgcn_tensor_load_to_lds)
#define HAVE_TDM 1
typedef __attribute__((ext_vector_type(4))) unsigned int v4u;
typedef __attribute__((ext_vector_type(8))) int v8i_t;
typedef __attribute__((ext_vector_type(4))) int v4i_t;

// 1D copy as a 1-row 2D tile: data_size=8B, tile_dim0=nqwords (<=65535),
// tile_dim1=1. D# layout per CDNA5 ISA ch.8 (group0: count/lds/global/type=2;
// group1: data_size, dims, strides; remaining groups zero = dims unused).
// This toolchain's builtin takes 6 args (g0 4xu32, g1 8xi32, 4xi32, 4xi32,
// 8xi32, cpol).
__device__ __forceinline__ void tdm_load_1d(const void* gsrc,
                                            unsigned lds_byte_addr,
                                            unsigned nqwords) {
  unsigned long long ga = (unsigned long long)gsrc;
  v4u g0;
  g0[0] = 1u;                                       // count=1, user, no gather
  g0[1] = lds_byte_addr;                            // lds_addr [63:32]
  g0[2] = (unsigned)ga;                             // global_addr lo
  g0[3] = (unsigned)((ga >> 32) & 0x01ffffffu) | 0x80000000u;  // hi | type=2
  v8i_t g1;
  g1[0] = 3 << 16;                                  // data_size=8B, mask=0
  g1[1] = (int)((nqwords & 0xffffu) << 16);         // tensor_dim0[15:0]
  g1[2] = (int)((nqwords >> 16) & 0xffffu) | (1 << 16);  // dim0 hi | dim1=1
  g1[3] = (int)((nqwords & 0xffffu) << 16);         // tile_dim0
  g1[4] = 1;                                        // tile_dim1=1, tile_dim2=0
  g1[5] = (int)nqwords;                             // dim0_stride lo
  g1[6] = 0;
  g1[7] = 0;
  v4i_t z4 = {0, 0, 0, 0};
  v8i_t z8 = {0, 0, 0, 0, 0, 0, 0, 0};
  __builtin_amdgcn_tensor_load_to_lds(g0, g1, z4, z4, z8, 0);
}
#else
#define HAVE_TDM 0
#endif

// ---------------- Kernel 0: pack all weights into bf16 fragments -------------
__device__ __forceinline__ void frag_pack(const float* __restrict__ src,
                                          __bf16* __restrict__ dst, int Kreal,
                                          int N, int nt, int total, int tid,
                                          int stride) {
  for (int i = tid; i < total; i += stride) {
    int e = i & 15, lane = (i >> 4) & 31, f = i >> 9;
    int t = f % nt, kb = f / nt;
    int k = kb * 32 + ((e & 8) << 1) + ((lane >> 4) << 3) + (e & 7);
    int n = t * 16 + (lane & 15);
    float v = (k < Kreal) ? src[k * N + n] : 0.f;
    dst[i] = (__bf16)v;
  }
}

__global__ void prep_weights(const float* saw0, const float* saw1,
                             const float* saw2, const float* fpw0,
                             const float* fpw1, const float* c1w,
                             __bf16* dst) {
  const int tid = blockIdx.x * blockDim.x + threadIdx.x;
  const int stride = gridDim.x * blockDim.x;
  __bf16* d = dst;
  for (int s = 0; s < 4; ++s)  // sa_w0: (6,128) padded to K=32
    frag_pack(saw0 + s * 6 * 128, d + s * (FR_SAW0 / 4), 6, 128, 8,
              FR_SAW0 / 4, tid, stride);
  d += FR_SAW0;
  for (int s = 0; s < 4; ++s)
    frag_pack(saw1 + s * 128 * 128, d + s * (FR_SAW1 / 4), 128, 128, 8,
              FR_SAW1 / 4, tid, stride);
  d += FR_SAW1;
  for (int s = 0; s < 4; ++s)
    frag_pack(saw2 + s * 128 * 256, d + s * (FR_SAW2 / 4), 128, 256, 16,
              FR_SAW2 / 4, tid, stride);
  d += FR_SAW2;
  frag_pack(fpw0, d, 1024, 128, 8, FR_FPW0, tid, stride);
  d += FR_FPW0;
  frag_pack(fpw1, d, 128, 64, 4, FR_FPW1, tid, stride);
  d += FR_FPW1;
  frag_pack(c1w, d, 64, 64, 4, FR_C1W, tid, stride);
}

// ---------------- Kernel 1: farthest point sampling (1 block) ----------------
__global__ __launch_bounds__(1024) void fps_kernel(const float* __restrict__ xyz,
                                                   int* fidx, float* newxyz) {
  __shared__ float rv[1024];
  __shared__ int ri[1024];
  __shared__ int s_far;
  __shared__ float s_c[3];
  const int tid = threadIdx.x;
  float px[16], py[16], pz[16], md[16];
#pragma unroll
  for (int k = 0; k < 16; ++k) {
    int p = tid + 1024 * k;
    px[k] = xyz[p * 3 + 0];
    py[k] = xyz[p * 3 + 1];
    pz[k] = xyz[p * 3 + 2];
    md[k] = 1e10f;
  }
  if (tid == 0) s_far = 0;
  __syncthreads();
  for (int it = 0; it < NCTR; ++it) {
    int far = s_far;
    if (tid == 0) {
      fidx[it] = far;
      float cx = xyz[far * 3], cy = xyz[far * 3 + 1], cz = xyz[far * 3 + 2];
      newxyz[it * 3] = cx; newxyz[it * 3 + 1] = cy; newxyz[it * 3 + 2] = cz;
      s_c[0] = cx; s_c[1] = cy; s_c[2] = cz;
    }
    __syncthreads();
    float cx = s_c[0], cy = s_c[1], cz = s_c[2];
    float bv = -1.f;
    int bi = 0x7fffffff;
#pragma unroll
    for (int k = 0; k < 16; ++k) {
      float dx = px[k] - cx, dy = py[k] - cy, dz = pz[k] - cz;
      float d2 = dx * dx + dy * dy + dz * dz;
      md[k] = fminf(md[k], d2);
      if (md[k] > bv) { bv = md[k]; bi = tid + 1024 * k; }
    }
    rv[tid] = bv; ri[tid] = bi;
    __syncthreads();
    for (int s = 512; s > 0; s >>= 1) {
      if (tid < s) {
        if (rv[tid + s] > rv[tid] ||
            (rv[tid + s] == rv[tid] && ri[tid + s] < ri[tid])) {
          rv[tid] = rv[tid + s];
          ri[tid] = ri[tid + s];
        }
      }
      __syncthreads();
    }
    if (tid == 0) s_far = ri[0];
    __syncthreads();
  }
}

// ---------------- Kernel 2: ball query (1 wave per center x scale) -----------
__global__ __launch_bounds__(32) void ball_query_kernel(
    const float* __restrict__ xyz, const float* __restrict__ newxyz,
    int* __restrict__ ballidx) {
  const int ctr = blockIdx.x, s = blockIdx.y, lane = threadIdx.x;
  const int nsample = c_ns[s];
  const float r2 = c_r2[s];
  int* dst = ballidx + c_soff[s] + ctr * nsample;
  const float cx = newxyz[ctr * 3], cy = newxyz[ctr * 3 + 1],
              cz = newxyz[ctr * 3 + 2];
  int total = 0;
  for (int start = 0; start < NPTS && total < nsample; start += 32) {
    int p = start + lane;
    float dx = xyz[p * 3] - cx, dy = xyz[p * 3 + 1] - cy,
          dz = xyz[p * 3 + 2] - cz;
    bool pred = (dx * dx + dy * dy + dz * dz) <= r2;
    unsigned mm = (unsigned)__ballot(pred);
    int off = __popc(mm & ((1u << lane) - 1u));
    int pos = total + off;
    if (pred && pos < nsample) dst[pos] = p;
    total += __popc(mm);
  }
  if (total > nsample) total = nsample;
  int first = NPTS;  // pad value (gather clamps), matches empty-ball semantics
  if (total > 0) first = dst[0];
  for (int q = total + lane; q < nsample; q += 32) dst[q] = first;
}

// ---------------- Kernel 3: per-scale grouped MLP + maxpool (WMMA) -----------
// Per-scale weight fragments staged in dynamic LDS (104 KB) via the Tensor
// Data Mover and reused by 4 waves x up to 256 chunks; activations in static
// LDS (A-swizzled).
__global__ __launch_bounds__(128) void sa_mlp_kernel(
    const float* __restrict__ xyz, const float* __restrict__ feats,
    const float* __restrict__ newxyz, const int* __restrict__ ballidx,
    const __bf16* __restrict__ wsbf, const float* __restrict__ b0,
    const float* __restrict__ b1, const float* __restrict__ b2,
    float* __restrict__ l1feat) {
  extern __shared__ __align__(32) __bf16 lW[];     // dynamic: weight frags
  __shared__ alignas(32) __bf16 sAin[4][512];      // K=32 A-swizzled
  __shared__ alignas(32) __bf16 sH0[4][16 * 128];  // K=128 A-swizzled
  __shared__ alignas(32) __bf16 sH1[4][16 * 128];
  __shared__ int smax[256];
  const int ctr = blockIdx.x, s = blockIdx.y;
  const int tid = threadIdx.x, wave = tid >> 5, lane = tid & 31;
  const int nsample = c_ns[s];
  const int nchunk = nsample >> 4;
  const int* bidx = ballidx + c_soff[s] + ctr * nsample;
  const __bf16* W0f = wsbf + s * (FR_SAW0 / 4);
  const __bf16* W1f = wsbf + FR_SAW0 + s * (FR_SAW1 / 4);
  const __bf16* W2f = wsbf + FR_SAW0 + FR_SAW1 + s * (FR_SAW2 / 4);
  __bf16* lW0 = lW;                                // 4096 elems (8 frags)
  __bf16* lW1 = lW0 + FR_SAW0 / 4;                 // 16384 elems (32 frags)
  __bf16* lW2 = lW1 + FR_SAW1 / 4;                 // 32768 elems (64 frags)
  const float* bb0 = b0 + s * 128;
  const float* bb1 = b1 + s * 128;
  const float* bb2 = b2 + s * 256;
  const float cx = newxyz[ctr * 3], cy = newxyz[ctr * 3 + 1],
              cz = newxyz[ctr * 3 + 2];

  // Stage this scale's weight fragments into LDS.
#if HAVE_TDM
  if (tid < 32) {  // one wave issues the DMA; EXEC-independent, TENSORcnt
    unsigned gss = __builtin_amdgcn_groupstaticsize();  // dynamic-LDS base
    tdm_load_1d(W0f, gss, (FR_SAW0 / 4) * 2 / 8);
    tdm_load_1d(W1f, gss + (FR_SAW0 / 4) * 2, (FR_SAW1 / 4) * 2 / 8);
    tdm_load_1d(W2f, gss + (FR_SAW0 / 4 + FR_SAW1 / 4) * 2,
                (FR_SAW2 / 4) * 2 / 8);
    __builtin_amdgcn_s_wait_tensorcnt(0);
  }
#else
  {
    const uint4* g0 = (const uint4*)W0f;
    const uint4* g1 = (const uint4*)W1f;
    const uint4* g2 = (const uint4*)W2f;
    uint4* l0 = (uint4*)lW0;
    uint4* l1 = (uint4*)lW1;
    uint4* l2 = (uint4*)lW2;
    for (int i = tid; i < (FR_SAW0 / 4) / 8; i += 128) l0[i] = g0[i];
    for (int i = tid; i < (FR_SAW1 / 4) / 8; i += 128) l1[i] = g1[i];
    for (int i = tid; i < (FR_SAW2 / 4) / 8; i += 128) l2[i] = g2[i];
  }
#endif
  // Preload per-lane biases (loop-invariant; atomics below defeat LICM).
  float bs0[8], bs1[8], bs2[16];
  {
    int n = lane & 15;
#pragma unroll
    for (int t = 0; t < 8; ++t) bs0[t] = bb0[t * 16 + n];
#pragma unroll
    for (int t = 0; t < 8; ++t) bs1[t] = bb1[t * 16 + n];
#pragma unroll
    for (int t = 0; t < 16; ++t) bs2[t] = bb2[t * 16 + n];
  }
  for (int k = tid; k < 256; k += 128) smax[k] = 0;
  __syncthreads();

  for (int c = wave; c < nchunk; c += 4) {
    // Build 16x6 input tile (xyz-delta | feature) in A-swizzled layout.
    // Channel r<8 maps to offset r; zeroing offsets 6..31 zeroes K=6..31.
    if (lane < 16) {
      int j = c * 16 + lane;
      int pi = bidx[j];
      if (pi > NPTS - 1) pi = NPTS - 1;
      __bf16* row = &sAin[wave][lane * 32];
      row[0] = (__bf16)(xyz[pi * 3 + 0] - cx);
      row[1] = (__bf16)(xyz[pi * 3 + 1] - cy);
      row[2] = (__bf16)(xyz[pi * 3 + 2] - cz);
      row[3] = (__bf16)(feats[0 * NPTS + pi]);
      row[4] = (__bf16)(feats[1 * NPTS + pi]);
      row[5] = (__bf16)(feats[2 * NPTS + pi]);
#pragma unroll
      for (int q = 6; q < 32; ++q) row[q] = (__bf16)0.f;
    }
    // Layer 0: 6(->32) -> 128
    {
      v16bf a = load_a_frag(sAin[wave], 0, lane);
#pragma unroll
      for (int t = 0; t < 8; ++t) {
        v8f acc = splat8(bs0[t]);
        v16bf b = load_b_frag(lW0, 8, 0, t, lane);
        acc = wmma_bf16(a, b, acc);
        store_c_relu_af(sH0[wave], t * 16, acc, lane);
      }
    }
    // Layer 1: 128 -> 128
#pragma unroll
    for (int t = 0; t < 8; ++t) {
      v8f acc = splat8(bs1[t]);
#pragma unroll
      for (int kb = 0; kb < 4; ++kb) {
        v16bf a = load_a_frag(sH0[wave], kb, lane);
        v16bf b = load_b_frag(lW1, 8, kb, t, lane);
        acc = wmma_bf16(a, b, acc);
      }
      store_c_relu_af(sH1[wave], t * 16, acc, lane);
    }
    // Layer 2: 128 -> 256, ReLU, max-pool over the 16 rows.
#pragma unroll
    for (int t = 0; t < 16; ++t) {
      v8f acc = splat8(bs2[t]);
#pragma unroll
      for (int kb = 0; kb < 4; ++kb) {
        v16bf a = load_a_frag(sH1[wave], kb, lane);
        v16bf b = load_b_frag(lW2, 16, kb, t, lane);
        acc = wmma_bf16(a, b, acc);
      }
      float m = 0.f;
#pragma unroll
      for (int j = 0; j < 8; ++j) {
        float v = acc[j];
        v = v > 0.f ? v : 0.f;
        m = fmaxf(m, v);
      }
      m = fmaxf(m, __shfl_xor(m, 16, 32));
      // post-ReLU values >= 0 -> ordered-int max is exact for float
      atomicMax(&smax[t * 16 + (lane & 15)], __float_as_int(m));
    }
  }
  __syncthreads();
  for (int k = tid; k < 256; k += 128)
    l1feat[ctr * 1024 + s * 256 + k] = __int_as_float(smax[k]);
}

// ---------------- Kernel 4: 3-NN among centers + interp weights --------------
__global__ void nn_kernel(const float* __restrict__ xyz,
                          const float* __restrict__ newxyz,
                          int* __restrict__ nni, float* __restrict__ nnw) {
  int p = blockIdx.x * blockDim.x + threadIdx.x;
  if (p >= NPTS) return;
  float x = xyz[p * 3], y = xyz[p * 3 + 1], z = xyz[p * 3 + 2];
  float d0 = 1e30f, d1 = 1e30f, d2v = 1e30f;
  int i0 = 0, i1 = 0, i2 = 0;
  for (int c = 0; c < NCTR; ++c) {
    float dx = x - newxyz[c * 3], dy = y - newxyz[c * 3 + 1],
          dz = z - newxyz[c * 3 + 2];
    float d = dx * dx + dy * dy + dz * dz;
    if (d < d0)      { d2v = d1; i2 = i1; d1 = d0; i1 = i0; d0 = d; i0 = c; }
    else if (d < d1) { d2v = d1; i2 = i1; d1 = d; i1 = c; }
    else if (d < d2v){ d2v = d; i2 = c; }
  }
  float w0 = 1.f / (d0 + 1e-8f), w1 = 1.f / (d1 + 1e-8f),
        w2 = 1.f / (d2v + 1e-8f);
  float sw = w0 + w1 + w2;
  w0 /= sw; w1 /= sw; w2 /= sw;
  nni[p * 3] = i0; nni[p * 3 + 1] = i1; nni[p * 3 + 2] = i2;
  nnw[p * 3] = w0; nnw[p * 3 + 1] = w1; nnw[p * 3 + 2] = w2;
}

// ---------------- Kernel 5: interp + FP MLPs + head (WMMA) -------------------
__global__ __launch_bounds__(128) void fp_kernel(
    const float* __restrict__ l1feat, const int* __restrict__ nni,
    const float* __restrict__ nnw, const __bf16* __restrict__ wsbf,
    const float* __restrict__ fpb0, const float* __restrict__ fpb1,
    const float* __restrict__ c1b, const float* __restrict__ bng,
    const float* __restrict__ bnb, const float* __restrict__ c2w,
    const float* __restrict__ c2b, float* __restrict__ out) {
  __shared__ alignas(32) __bf16 sA[4][512];          // one 32-wide K block
  __shared__ alignas(32) __bf16 sH1[4][16 * 128];
  __shared__ alignas(32) __bf16 sH2[4][16 * 64];
  __shared__ float sH3[4][16 * 64];
  const int tid = threadIdx.x, wave = tid >> 5, lane = tid & 31;
  const int tile = blockIdx.x * 4 + wave;  // 0..1023 (16 rows each)
  const __bf16* FW0f = wsbf + FR_SAW0 + FR_SAW1 + FR_SAW2;
  const __bf16* FW1f = FW0f + FR_FPW0;
  const __bf16* C1Wf = FW1f + FR_FPW1;
  __builtin_prefetch(FW0f, 0, 1);

  const int r = lane & 15;
  const int chh = lane >> 4;
  const int p = tile * 16 + r;
  const int n0 = nni[p * 3], n1 = nni[p * 3 + 1], n2 = nni[p * 3 + 2];
  const float w0 = nnw[p * 3], w1 = nnw[p * 3 + 1], w2 = nnw[p * 3 + 2];

  // GEMM: interp[16x1024] x fpW0[1024x128], K streamed in 32-wide blocks.
  v8f acc[8];
#pragma unroll
  for (int t = 0; t < 8; ++t) acc[t] = bias_frag(fpb0, t * 16, lane);
  for (int kb = 0; kb < 32; ++kb) {
    int cb = kb * 32 + chh * 16;
    __bf16* base = &sA[wave][r * 32];
#pragma unroll
    for (int i = 0; i < 16; ++i) {
      int ch = cb + i;
      float v = w0 * l1feat[n0 * 1024 + ch] + w1 * l1feat[n1 * 1024 + ch] +
                w2 * l1feat[n2 * 1024 + ch];
      base[aswz(chh * 16 + i)] = (__bf16)v;
    }
    v16bf a = load_a_frag(sA[wave], 0, lane);
#pragma unroll
    for (int t = 0; t < 8; ++t) {
      v16bf b = load_b_frag(FW0f, 8, kb, t, lane);
      acc[t] = wmma_bf16(a, b, acc[t]);
    }
  }
#pragma unroll
  for (int t = 0; t < 8; ++t) store_c_relu_af(sH1[wave], t * 16, acc[t], lane);

  // fp1: 128 -> 64
#pragma unroll
  for (int t = 0; t < 4; ++t) {
    v8f a2 = bias_frag(fpb1, t * 16, lane);
#pragma unroll
    for (int kb = 0; kb < 4; ++kb) {
      v16bf a = load_a_frag(sH1[wave], kb, lane);
      v16bf b = load_b_frag(FW1f, 4, kb, t, lane);
      a2 = wmma_bf16(a, b, a2);
    }
    store_c_relu_af(sH2[wave], t * 16, a2, lane);
  }

  // conv1: 64 -> 64, then BN scale/shift + leaky ReLU (f32 in LDS, row-major)
  const float inv = rsqrtf(1.f + 1e-5f);
#pragma unroll
  for (int t = 0; t < 4; ++t) {
    v8f a3 = bias_frag(c1b, t * 16, lane);
#pragma unroll
    for (int kb = 0; kb < 2; ++kb) {
      v16bf a = load_a_frag(sH2[wave], kb, lane);
      v16bf b = load_b_frag(C1Wf, 4, kb, t, lane);
      a3 = wmma_bf16(a, b, a3);
    }
    int col = t * 16 + (lane & 15);
    float sc = bng[col] * inv, bt = bnb[col];
    int h = lane >> 4;
#pragma unroll
    for (int j = 0; j < 8; ++j) {
      float v = a3[j] * sc + bt;
      v = v > 0.f ? v : 0.01f * v;
      sH3[wave][(j + 8 * h) * 64 + col] = v;
    }
  }

  // conv2: 64 -> 1; log_softmax over a size-1 axis is identically 0, then
  // sigmoid. Written as x - x so the chain is preserved (not foldable
  // without fast-math) yet numerically exact.
  if (lane < 16) {
    float x = c2b[0];
#pragma unroll
    for (int c = 0; c < 64; ++c) x += sH3[wave][r * 64 + c] * c2w[c];
    float y = x - x;
    out[p] = 1.f / (1.f + expf(-y));
  }
}

// ---------------- Host launcher ----------------
extern "C" void kernel_launch(void* const* d_in, const int* in_sizes, int n_in,
                              void* d_out, int out_size, void* d_ws,
                              size_t ws_size, hipStream_t stream) {
  const float* pc    = (const float*)d_in[0];
  const float* feats = (const float*)d_in[1];
  const float* saw0  = (const float*)d_in[2];
  const float* sab0  = (const float*)d_in[3];
  const float* saw1  = (const float*)d_in[4];
  const float* sab1  = (const float*)d_in[5];
  const float* saw2  = (const float*)d_in[6];
  const float* sab2  = (const float*)d_in[7];
  const float* fpw0  = (const float*)d_in[8];
  const float* fpb0  = (const float*)d_in[9];
  const float* fpw1  = (const float*)d_in[10];
  const float* fpb1  = (const float*)d_in[11];
  const float* c1w   = (const float*)d_in[12];
  const float* c1b   = (const float*)d_in[13];
  const float* bng   = (const float*)d_in[14];
  const float* bnb   = (const float*)d_in[15];
  const float* c2w   = (const float*)d_in[16];
  const float* c2b   = (const float*)d_in[17];

  char* ws = (char*)d_ws;
  size_t off = 0;
  auto alignup = [](size_t x) { return (x + 255) & ~(size_t)255; };
  __bf16* wsbf = (__bf16*)(ws + off);
  off = alignup(off + (size_t)FR_TOTAL * 2);
  int* fidx = (int*)(ws + off);
  off = alignup(off + 128 * sizeof(int));
  float* newxyz = (float*)(ws + off);
  off = alignup(off + 128 * 3 * sizeof(float));
  int* ballidx = (int*)(ws + off);
  off = alignup(off + (size_t)128 * 5440 * sizeof(int));
  float* l1feat = (float*)(ws + off);
  off = alignup(off + (size_t)128 * 1024 * sizeof(float));
  int* nni = (int*)(ws + off);
  off = alignup(off + (size_t)NPTS * 3 * sizeof(int));
  float* nnw = (float*)(ws + off);
  off = alignup(off + (size_t)NPTS * 3 * sizeof(float));
  (void)ws_size; (void)n_in; (void)in_sizes; (void)out_size;

  prep_weights<<<512, 256, 0, stream>>>(saw0, saw1, saw2, fpw0, fpw1, c1w,
                                        wsbf);
  fps_kernel<<<1, 1024, 0, stream>>>(pc, fidx, newxyz);
  ball_query_kernel<<<dim3(128, 4), 32, 0, stream>>>(pc, newxyz, ballidx);
  sa_mlp_kernel<<<dim3(128, 4), 128, SA_LDS_W_BYTES, stream>>>(
      pc, feats, newxyz, ballidx, wsbf, sab0, sab1, sab2, l1feat);
  nn_kernel<<<64, 256, 0, stream>>>(pc, newxyz, nni, nnw);
  fp_kernel<<<256, 128, 0, stream>>>(l1feat, nni, nnw, wsbf, fpb0, fpb1, c1b,
                                     bng, bnb, c2w, c2b, (float*)d_out);
}
